// SNN_65025804862120
// MI455X (gfx1250) — matrix operand order repaired and checked by reading, not compiled, not run on hardware
//
#include <hip/hip_runtime.h>

// Fused SNN forward for MI455X (gfx1250, wave32, WMMA):
//   spikes[512,128,256] --(w_h, f16 WMMA)--> CUBA-LIF --(w_o, f16 WMMA)--> CUBA-LI -> out[512,128,32]
// Kernel 1 (once per launch): convert w_h f32 -> f16, pre-swizzled into per-lane
//   WMMA B-fragment order in d_ws (512 KB, L2-resident).
// Kernel 2: one workgroup per 16-batch tile, 32 waves x 32 hidden neurons.
//   Recurrent state in VGPRs, z exchanged through LDS, B fragments streamed
//   from L2 as contiguous 32B/lane loads. An opaque scalar offset (inline asm)
//   keeps those loads inside the time loop so LICM cannot hoist-and-spill them.

typedef __attribute__((ext_vector_type(16))) _Float16 v16h;
typedef __attribute__((ext_vector_type(8)))  _Float16 v8h;
typedef __attribute__((ext_vector_type(4)))  _Float16 v4h;
typedef __attribute__((ext_vector_type(8)))  float    v8f;
typedef __attribute__((ext_vector_type(4)))  float    v4f;

#define T_STEPS 512
#define BATCH   128
#define N_IN    256
#define N_H     1024
#define N_OUT   32
#define BT      16      // batch tile = WMMA M
#define DECAY   0.875f  // 1 - dt/tau
#define DVF     0.125f  // dt/tau
#define FRAG_H  512     // halves per fragment-slot (32 lanes x 16)

// ---------------- prep: w_h -> f16 B-fragments in ws ----------------
// Layout: whB[(tile*8 + kc)*512 + lane*16 + j]
//   tile = h/16 (64 tiles), kc = K chunk of 32 (8 chunks)
//   B-fragment per lane: n = lane&15 selects column (h row of w_h),
//   16 consecutive K values starting at kc*32 + (lane<16 ? 0 : 16).
__global__ void prep_wh_kernel(const float* __restrict__ w_h,
                               _Float16* __restrict__ whB) {
  int idx  = blockIdx.x * blockDim.x + threadIdx.x;  // 0 .. 16383
  int lane = idx & 31;
  int kc   = (idx >> 5) & 7;
  int tile = idx >> 8;
  const float* src = w_h + (size_t)(tile * 16 + (lane & 15)) * N_IN +
                     kc * 32 + ((lane >> 4) & 1) * 16;
  v16h r;
#pragma unroll
  for (int j = 0; j < 16; ++j) r[j] = (_Float16)src[j];
  *(v16h*)(whB + (size_t)idx * 16) = r;
}

// A-fragment (contiguous k order used here, matches B): halves 0-15 = K[kb..kb+15]
__device__ __forceinline__ v16h ldsA(const _Float16* row, int col) {
  v8h lo = *(const v8h*)(row + col);
  v8h hi = *(const v8h*)(row + col + 8);
  v16h r;
#pragma unroll
  for (int j = 0; j < 8; ++j) { r[j] = lo[j]; r[j + 8] = hi[j]; }
  return r;
}

// B-fragment (contiguous k order used here, matches A): 16 consecutive K values per lane-half
__device__ __forceinline__ v16h ldsB(const _Float16* row, int col) {
  v8h lo = *(const v8h*)(row + col);
  v8h hi = *(const v8h*)(row + col + 8);
  v16h r;
#pragma unroll
  for (int j = 0; j < 8; ++j) { r[j] = lo[j]; r[j + 8] = hi[j]; }
  return r;
}

__device__ __forceinline__ v8f wmma(v16h a, v16h b, v8f c) {
  return __builtin_amdgcn_wmma_f32_16x16x32_f16(false, a, false, b, (short)0, c,
                                                false, false);
}

// ---------------- fused recurrent kernel ----------------
__global__ __launch_bounds__(1024) void snn_fused_kernel(
    const float* __restrict__ spikes, const _Float16* __restrict__ whB,
    const float* __restrict__ w_o, float* __restrict__ out) {
  __shared__ _Float16 sSpk[BT * N_IN];    //  8 KB: f16 spike tile for this t
  __shared__ _Float16 sZ[BT * N_H];       // 32 KB: f16 hidden spikes for this t
  __shared__ _Float16 sWo[N_OUT * N_H];   // 64 KB: f16 readout weights (one-time)

  const int tid    = threadIdx.x;
  const int lane   = tid & 31;
  const int wv     = tid >> 5;            // 32 waves
  const int bt     = blockIdx.x;          // 8 batch tiles
  const int n      = lane & 15;
  const int hiHalf = (lane >> 4) & 1;     // 0 = lanes 0-15, 1 = lanes 16-31
  const int kbA    = hiHalf * 16;         // A-fragment K base
  const int kbB    = hiHalf * 16;         // B-fragment K base

  // --- one-time: stage w_o (f32 -> f16) into LDS ---
  for (int idx = tid * 4; idx < N_OUT * N_H; idx += 1024 * 4) {
    v4f f = *(const v4f*)(w_o + idx);
    v4h h;
    h[0] = (_Float16)f.x; h[1] = (_Float16)f.y;
    h[2] = (_Float16)f.z; h[3] = (_Float16)f.w;
    *(v4h*)(sWo + idx) = h;
  }

  // recurrent state, register-resident for the whole trajectory
  v8f vm0 = {}, vm1 = {}, is0 = {}, is1 = {};  // hidden LIF (2 tiles)
  v8f vo = {}, io = {};                        // readout LI (waves 0,1)

  // pre-swizzled B fragments for this wave's two h-tiles (contiguous 32B/lane)
  const _Float16* wb0 = whB + (size_t)((wv * 2 + 0) * 8) * FRAG_H + lane * 16;
  const _Float16* wb1 = whB + (size_t)((wv * 2 + 1) * 8) * FRAG_H + lane * 16;

  const _Float16* aSpkRow = sSpk + n * N_IN;
  const _Float16* aZRow   = sZ + n * N_H;
  const _Float16* bWoRow  = sWo + (wv * 16 + n) * N_H;  // used only by wv<2

  const size_t tStride = (size_t)BATCH * N_IN;
  const float* spBase  = spikes + (size_t)bt * BT * N_IN;
  const int    hbase   = wv * 32 + n;

  for (int t = 0; t < T_STEPS; ++t) {
    const float* sp = spBase + (size_t)t * tStride;

    // stage spike tile: 16x256 f32 (contiguous) -> f16 LDS, 4 elems/thread
    {
      v4f f = *(const v4f*)(sp + tid * 4);
      v4h h;
      h[0] = (_Float16)f.x; h[1] = (_Float16)f.y;
      h[2] = (_Float16)f.z; h[3] = (_Float16)f.w;
      *(v4h*)(sSpk + tid * 4) = h;
    }
    if (t + 1 < T_STEPS)
      __builtin_prefetch(sp + tStride + tid * 16, 0, 3);  // next-t spike tile
    __syncthreads();

    // Opaque zero offset: the compiler cannot prove the B-fragment loads are
    // loop-invariant, so they stay in the loop (L2-hot streams) instead of
    // being hoisted into 128 VGPRs and spilled to scratch.
    int toff = 0;
    asm volatile("" : "+s"(toff));

    // --- hidden GEMM: c[16,32h] += S[16,256] * w_h^T, K in 8 chunks of 32 ---
    v8f c0 = {}, c1 = {};
#pragma unroll
    for (int kc = 0; kc < 8; ++kc) {
      v16h a  = ldsA(aSpkRow, kc * 32 + kbA);
      v16h b0 = *(const v16h*)(wb0 + kc * FRAG_H + toff);  // 2x global_load_b128
      v16h b1 = *(const v16h*)(wb1 + kc * FRAG_H + toff);
      c0 = wmma(a, b0, c0);
      c1 = wmma(a, b1, c1);
    }

    // --- CUBA-LIF update (elementwise on accumulator layout) ---
    is0 = is0 * DECAY + c0;
    is1 = is1 * DECAY + c1;
    vm0 = vm0 * DECAY + is0 * DVF;
    vm1 = vm1 * DECAY + is1 * DVF;
#pragma unroll
    for (int r = 0; r < 8; ++r) {
      int m = r + hiHalf * 8;  // batch row of this accumulator element
      float v0 = vm0[r], v1 = vm1[r];
      bool s0 = v0 > 1.0f, s1 = v1 > 1.0f;
      vm0[r] = s0 ? 0.0f : v0;  // reset-to-zero on spike
      vm1[r] = s1 ? 0.0f : v1;
      sZ[m * N_H + hbase]      = s0 ? (_Float16)1.0f : (_Float16)0.0f;
      sZ[m * N_H + hbase + 16] = s1 ? (_Float16)1.0f : (_Float16)0.0f;
    }
    __syncthreads();

    // --- readout GEMM + LI: waves 0,1 each own one 16x16 output tile, K=1024 ---
    if (wv < 2) {
      v8f c = {};
#pragma unroll
      for (int kc = 0; kc < 32; ++kc) {
        v16h a = ldsA(aZRow, kc * 32 + kbA);
        v16h b = ldsB(bWoRow, kc * 32 + kbB);
        c = wmma(a, b, c);
      }
      io = io * DECAY + c;
      vo = vo * DECAY + io * DVF;
      float* op = out + (size_t)t * (BATCH * N_OUT) +
                  (size_t)bt * BT * N_OUT + wv * 16 + n;
#pragma unroll
      for (int r = 0; r < 8; ++r) op[(r + hiHalf * 8) * N_OUT] = vo[r];
    }
  }
}

extern "C" void kernel_launch(void* const* d_in, const int* in_sizes, int n_in,
                              void* d_out, int out_size, void* d_ws,
                              size_t ws_size, hipStream_t stream) {
  const float* spikes = (const float*)d_in[0];
  const float* w_h    = (const float*)d_in[1];
  const float* w_o    = (const float*)d_in[2];
  float* out          = (float*)d_out;
  _Float16* whB       = (_Float16*)d_ws;  // 512 KB of fragment-ordered f16 w_h
  (void)in_sizes; (void)n_in; (void)out_size; (void)ws_size;

  // 64 tiles * 8 chunks * 32 lanes = 16384 fragment-lanes
  prep_wh_kernel<<<dim3(64), dim3(256), 0, stream>>>(w_h, whB);
  snn_fused_kernel<<<dim3(BATCH / BT), dim3(1024), 0, stream>>>(spikes, whB,
                                                                w_o, out);
}
